// CNNMotionEstimator_69776038691204
// MI455X (gfx1250) — matrix operand order repaired and checked by reading, not compiled, not run on hardware
//
#include <hip/hip_runtime.h>
#include <hip/hip_bf16.h>

typedef __attribute__((ext_vector_type(16))) __bf16 v16bf;
typedef __attribute__((ext_vector_type(8)))  float  v8f;
typedef unsigned int  u32;
typedef unsigned short u16;

// ---------- bf16 helpers (round-to-nearest-even, no __bf16 arithmetic) ----------
__device__ __forceinline__ __bf16 f2bf(float f) {
    union { float f; u32 u; } c; c.f = f;
    u32 r = (c.u + 0x7FFFu + ((c.u >> 16) & 1u)) >> 16;
    union { u16 s; __bf16 b; } o; o.s = (u16)r; return o.b;
}

// =====================================================================
// Weight layout ("swizzled B"): for each 32-wide K tile, 32 consecutive
// k-values of one output column stored contiguously:
//   W'[ ((k>>5)*N + n)*32 + (k&31) ]
// A lane of v_wmma_f32_16x16x32_bf16 (col n, k = k0 + half*8 + 0..7 and k0 + 16 + half*8 + 0..7)
// then loads its whole B fragment as two 16-byte pieces at byte offsets 0 and 32 (2x b128).
// =====================================================================

// ---------- 1x1-tile GEMM (small FC heads): out[M,N](f32) ----------
__global__ void gemm_wmma_bf16(const __bf16* __restrict__ A,
                               const __bf16* __restrict__ Wm,
                               const float* __restrict__ bias,
                               float* __restrict__ out,
                               int M, int K, int N, int relu)
{
    int wave = (int)((blockIdx.x * blockDim.x + threadIdx.x) >> 5);
    int lane = threadIdx.x & 31;
    int tilesN = N >> 4, tilesM = M >> 4;
    if (wave >= tilesM * tilesN) return;
    int tm = wave % tilesM, tn = wave / tilesM;
    int half = lane >> 4, mn = lane & 15;
    int n = tn * 16 + mn;

    const __bf16* Arow = A + (size_t)(tm * 16 + mn) * K + half * 8;
    const __bf16* wp   = Wm + (size_t)n * 32 + half * 8;
    const size_t wstep = (size_t)N * 32;

    v8f acc = {};
    union AV { v16bf v; uint4 q[2]; } a, b;
    for (int k0 = 0; k0 < K; k0 += 32) {
        a.q[0] = *(const uint4*)(Arow + k0);
        a.q[1] = *(const uint4*)(Arow + k0 + 16);
        b.q[0] = *(const uint4*)(wp);
        b.q[1] = *(const uint4*)(wp + 16);
        wp += wstep;
        acc = __builtin_amdgcn_wmma_f32_16x16x32_bf16(false, a.v, false, b.v,
                                                      (short)0, acc, false, false);
    }
    float bs = bias ? bias[n] : 0.0f;
#pragma unroll
    for (int r = 0; r < 8; ++r) {
        float v = acc[r] + bs;
        if (relu) v = fmaxf(v, 0.0f);
        out[(size_t)(tm * 16 + r + half * 8) * N + n] = v;
    }
}

// ---------- 2x2 register-blocked GEMM: wave computes 32x32 of out ----------
// M%32==0, N%32==0, K%32==0. 4 WMMAs per 8 b128 loads per K-step.
__global__ void gemm_wmma_bf16_2x2(const __bf16* __restrict__ A,
                                   const __bf16* __restrict__ Wm,
                                   const float* __restrict__ bias,
                                   float* __restrict__ out,
                                   int M, int K, int N, int relu)
{
    int wave = (int)((blockIdx.x * blockDim.x + threadIdx.x) >> 5);
    int lane = threadIdx.x & 31;
    int tilesN = N >> 5, tilesM = M >> 5;
    if (wave >= tilesM * tilesN) return;
    int tm = wave % tilesM, tn = wave / tilesM;
    int half = lane >> 4, mn = lane & 15;
    int n0 = tn * 32 + mn;

    const __bf16* Ar0 = A + (size_t)(tm * 32 + mn) * K + half * 8;
    const __bf16* Ar1 = Ar0 + (size_t)16 * K;
    const __bf16* wp  = Wm + (size_t)n0 * 32 + half * 8;
    const size_t wstep = (size_t)N * 32;

    v8f c00 = {}, c01 = {}, c10 = {}, c11 = {};
    union AV { v16bf v; uint4 q[2]; } a0, a1, b0, b1;
    for (int k0 = 0; k0 < K; k0 += 32) {
        a0.q[0] = *(const uint4*)(Ar0 + k0);
        a0.q[1] = *(const uint4*)(Ar0 + k0 + 16);
        a1.q[0] = *(const uint4*)(Ar1 + k0);
        a1.q[1] = *(const uint4*)(Ar1 + k0 + 16);
        b0.q[0] = *(const uint4*)(wp);
        b0.q[1] = *(const uint4*)(wp + 16);
        b1.q[0] = *(const uint4*)(wp + 512);    // column block n0+16
        b1.q[1] = *(const uint4*)(wp + 528);
        wp += wstep;
        c00 = __builtin_amdgcn_wmma_f32_16x16x32_bf16(false, a0.v, false, b0.v, (short)0, c00, false, false);
        c01 = __builtin_amdgcn_wmma_f32_16x16x32_bf16(false, a0.v, false, b1.v, (short)0, c01, false, false);
        c10 = __builtin_amdgcn_wmma_f32_16x16x32_bf16(false, a1.v, false, b0.v, (short)0, c10, false, false);
        c11 = __builtin_amdgcn_wmma_f32_16x16x32_bf16(false, a1.v, false, b1.v, (short)0, c11, false, false);
    }
    float bs0 = bias ? bias[n0] : 0.0f;
    float bs1 = bias ? bias[n0 + 16] : 0.0f;
#pragma unroll
    for (int r = 0; r < 8; ++r) {
        int m0 = tm * 32 + r + half * 8;
        float v00 = c00[r] + bs0, v01 = c01[r] + bs1;
        float v10 = c10[r] + bs0, v11 = c11[r] + bs1;
        if (relu) { v00 = fmaxf(v00, 0.f); v01 = fmaxf(v01, 0.f);
                    v10 = fmaxf(v10, 0.f); v11 = fmaxf(v11, 0.f); }
        out[(size_t)m0 * N + n0]            = v00;
        out[(size_t)m0 * N + n0 + 16]       = v01;
        out[(size_t)(m0 + 16) * N + n0]     = v10;
        out[(size_t)(m0 + 16) * N + n0 + 16]= v11;
    }
}

// ---------- 2x2 register-blocked implicit-GEMM conv, NHWC bf16 ----------
// K = kh*kw*Cin, Cin%32==0, Cout%32==0, (B*Ho*Wo)%32==0.
__global__ void conv_wmma(const __bf16* __restrict__ in,
                          const __bf16* __restrict__ wt,
                          const float* __restrict__ bias,
                          __bf16* __restrict__ out,
                          int Bn, int Hin, int Win, int Cin,
                          int Ho, int Wo, int Cout,
                          int stride, int pad, int kh, int kw, int relu)
{
    int wave = (int)((blockIdx.x * blockDim.x + threadIdx.x) >> 5);
    int lane = threadIdx.x & 31;
    int M = Bn * Ho * Wo;
    int tilesN = Cout >> 5, tilesM = M >> 5;
    if (wave >= tilesM * tilesN) return;
    int tm = wave % tilesM, tn = wave / tilesM;
    int half = lane >> 4, mn = lane & 15;
    int n0 = tn * 32 + mn;

    // two output pixels per lane (rows m and m+16 of the 32-row block)
    int p0 = tm * 32 + mn, p1 = p0 + 16;
    int bb0 = p0 / (Ho * Wo); int re0 = p0 - bb0 * Ho * Wo;
    int oy0 = re0 / Wo;       int ox0 = re0 - oy0 * Wo;
    int bb1 = p1 / (Ho * Wo); int re1 = p1 - bb1 * Ho * Wo;
    int oy1 = re1 / Wo;       int ox1 = re1 - oy1 * Wo;
    int iy0a = oy0 * stride - pad, ix0a = ox0 * stride - pad;
    int iy0b = oy1 * stride - pad, ix0b = ox1 * stride - pad;

    const __bf16* wp   = wt + (size_t)n0 * 32 + half * 8;
    const size_t wstep = (size_t)Cout * 32;

    v8f c00 = {}, c01 = {}, c10 = {}, c11 = {};
    union AV { v16bf v; uint4 q[2]; } a0, a1, b0, b1;
    uint4 z4 = {0u, 0u, 0u, 0u};

    for (int tap = 0; tap < kh * kw; ++tap) {
        int ky = tap / kw, kx = tap - ky * kw;
        int iyA = iy0a + ky, ixA = ix0a + kx;
        int iyB = iy0b + ky, ixB = ix0b + kx;
        bool in0 = (iyA >= 0 && iyA < Hin && ixA >= 0 && ixA < Win);
        bool in1 = (iyB >= 0 && iyB < Hin && ixB >= 0 && ixB < Win);
        const __bf16* ap0 = in; const __bf16* ap1 = in;
        if (in0) ap0 = in + ((((size_t)bb0 * Hin + iyA) * Win + ixA) * Cin + half * 8);
        if (in1) ap1 = in + ((((size_t)bb1 * Hin + iyB) * Win + ixB) * Cin + half * 8);
        for (int cc = 0; cc < Cin; cc += 32) {
            if (in0) { a0.q[0] = *(const uint4*)(ap0 + cc);
                       a0.q[1] = *(const uint4*)(ap0 + cc + 16); }
            else     { a0.q[0] = z4; a0.q[1] = z4; }
            if (in1) { a1.q[0] = *(const uint4*)(ap1 + cc);
                       a1.q[1] = *(const uint4*)(ap1 + cc + 16); }
            else     { a1.q[0] = z4; a1.q[1] = z4; }
            b0.q[0] = *(const uint4*)(wp);
            b0.q[1] = *(const uint4*)(wp + 16);
            b1.q[0] = *(const uint4*)(wp + 512);
            b1.q[1] = *(const uint4*)(wp + 528);
            __builtin_prefetch(wp + wstep, 0, 1);   // global_prefetch_b8
            wp += wstep;
            c00 = __builtin_amdgcn_wmma_f32_16x16x32_bf16(false, a0.v, false, b0.v, (short)0, c00, false, false);
            c01 = __builtin_amdgcn_wmma_f32_16x16x32_bf16(false, a0.v, false, b1.v, (short)0, c01, false, false);
            c10 = __builtin_amdgcn_wmma_f32_16x16x32_bf16(false, a1.v, false, b0.v, (short)0, c10, false, false);
            c11 = __builtin_amdgcn_wmma_f32_16x16x32_bf16(false, a1.v, false, b1.v, (short)0, c11, false, false);
        }
    }
    float bs0 = bias ? bias[n0] : 0.0f;
    float bs1 = bias ? bias[n0 + 16] : 0.0f;
#pragma unroll
    for (int r = 0; r < 8; ++r) {
        int m0 = tm * 32 + r + half * 8;
        float v00 = c00[r] + bs0, v01 = c01[r] + bs1;
        float v10 = c10[r] + bs0, v11 = c11[r] + bs1;
        if (relu) { v00 = fmaxf(v00, 0.f); v01 = fmaxf(v01, 0.f);
                    v10 = fmaxf(v10, 0.f); v11 = fmaxf(v11, 0.f); }
        out[(size_t)m0 * Cout + n0]             = f2bf(v00);
        out[(size_t)m0 * Cout + n0 + 16]        = f2bf(v01);
        out[(size_t)(m0 + 16) * Cout + n0]      = f2bf(v10);
        out[(size_t)(m0 + 16) * Cout + n0 + 16] = f2bf(v11);
    }
}

// ---------- LayerNorm over D=256, one block per row ----------
__global__ void layernorm_k(const float* __restrict__ x, const float* __restrict__ g,
                            const float* __restrict__ bp, float* __restrict__ y, int D)
{
    __shared__ float red[256];
    int row = blockIdx.x, t = threadIdx.x;
    float v = x[(size_t)row * D + t];
    red[t] = v; __syncthreads();
    for (int s = 128; s > 0; s >>= 1) { if (t < s) red[t] += red[t + s]; __syncthreads(); }
    float mean = red[0] / (float)D; __syncthreads();
    float d = v - mean; red[t] = d * d; __syncthreads();
    for (int s = 128; s > 0; s >>= 1) { if (t < s) red[t] += red[t + s]; __syncthreads(); }
    float var = red[0] / (float)D;
    y[(size_t)row * D + t] = d * rsqrtf(var + 1e-5f) * g[t] + bp[t];
}

// ---------- splat ----------
__global__ void zero_f32(float* __restrict__ p, size_t n) {
    size_t i = (size_t)blockIdx.x * blockDim.x + threadIdx.x;
    if (i < n) p[i] = 0.0f;
}

__global__ void splat_scatter(const float* __restrict__ fused,
                              const float* __restrict__ coords,
                              float* __restrict__ fm, float* __restrict__ wm)
{
    const int W = 128, H = 192;
    int p = blockIdx.x;            // (b,n) pair, 0..8191
    int d = threadIdx.x;           // channel
    int b = p >> 11;
    float cx = coords[(size_t)p * 2 + 0], cy = coords[(size_t)p * 2 + 1];
    float sx = fminf(fmaxf(cx / 512.0f * (W - 1), 0.0f), (float)(W - 1));
    float sy = fminf(fmaxf(cy / 384.0f * (H - 1), 0.0f), (float)(H - 1));
    bool valid = cx >= 0.0f;
    int x0 = (int)floorf(sx), y0 = (int)floorf(sy);
    int x1 = min(x0 + 1, W - 1), y1 = min(y0 + 1, H - 1);
    float wx = sx - (float)x0, wy = sy - (float)y0;
    float w0 = (1.f - wx) * (1.f - wy), w1 = (1.f - wx) * wy;
    float w2 = wx * (1.f - wy),         w3 = wx * wy;
    if (!valid) { w0 = w1 = w2 = w3 = 0.0f; }
    int base = b * H * W;
    int i0 = base + y0 * W + x0, i1 = base + y1 * W + x0;
    int i2 = base + y0 * W + x1, i3 = base + y1 * W + x1;
    float f = fused[(size_t)p * 256 + d];
    atomicAdd(&fm[(size_t)i0 * 256 + d], w0 * f);
    atomicAdd(&fm[(size_t)i1 * 256 + d], w1 * f);
    atomicAdd(&fm[(size_t)i2 * 256 + d], w2 * f);
    atomicAdd(&fm[(size_t)i3 * 256 + d], w3 * f);
    if (d == 0) {
        atomicAdd(&wm[i0], w0); atomicAdd(&wm[i1], w1);
        atomicAdd(&wm[i2], w2); atomicAdd(&wm[i3], w3);
    }
}

__global__ void splat_norm(const float* __restrict__ fm, const float* __restrict__ wm,
                           __bf16* __restrict__ act)
{
    size_t i = (size_t)blockIdx.x * blockDim.x + threadIdx.x;
    size_t pix = i >> 8;
    float w = fmaxf(wm[pix], 1e-8f);
    act[i] = f2bf(fm[i] / w);
}

// ---------- weight transforms (write swizzled B layout) ----------
__global__ void wconv_trans(const float* __restrict__ src, __bf16* __restrict__ dst,
                            int co, int ci, int kh, int kw, int CinP, int CoutP)
{
    int idx = (int)(blockIdx.x * blockDim.x + threadIdx.x);
    int total = kh * kw * CinP * CoutP;
    if (idx >= total) return;
    int n = idx % CoutP; int k = idx / CoutP;     // logical (k, n)
    int c = k % CinP;    int r = k / CinP;
    int ky = r / kw,     kx = r - ky * kw;
    float v = 0.0f;
    if (c < ci && n < co) v = src[((((size_t)n * ci + c) * kh + ky) * kw) + kx];
    dst[((size_t)(k >> 5) * CoutP + n) * 32 + (k & 31)] = f2bf(v);
}

__global__ void wfc_trans(const float* __restrict__ src, __bf16* __restrict__ dst,
                          int K, int N, int NP, int permC, int permS)
{
    int idx = (int)(blockIdx.x * blockDim.x + threadIdx.x);
    if (idx >= K * NP) return;
    int n = idx % NP; int k = idx / NP;
    float v = 0.0f;
    if (n < N) {
        int ks = k;
        if (permC > 0) { int c = k % permC; int s = k / permC; ks = c * permS + s; }
        v = src[(size_t)ks * N + n];
    }
    dst[((size_t)(k >> 5) * NP + n) * 32 + (k & 31)] = f2bf(v);
}

__global__ void pad_bias(const float* __restrict__ src, float* __restrict__ dst, int n, int np)
{
    int i = (int)(blockIdx.x * blockDim.x + threadIdx.x);
    if (i < np) dst[i] = (i < n) ? src[i] : 0.0f;
}

// ---------- misc prep ----------
__global__ void build_fuseA(const float* __restrict__ patch, const float* __restrict__ img,
                            __bf16* __restrict__ A)
{
    size_t idx = (size_t)blockIdx.x * blockDim.x + threadIdx.x; // 8192*512
    int k = (int)(idx & 511); size_t p = idx >> 9; int b = (int)(p >> 11);
    float v = (k < 256) ? patch[p * 256 + k] : img[(size_t)b * 256 + (k - 256)];
    A[idx] = f2bf(v);
}

__global__ void packA_bf16(const __bf16* __restrict__ src, __bf16* __restrict__ dst,
                           int rows, int K)
{
    size_t idx = (size_t)blockIdx.x * blockDim.x + threadIdx.x; // 16*K
    int m = (int)(idx / (size_t)K);
    if (m >= 16) return;
    dst[idx] = (m < rows) ? src[idx] : f2bf(0.0f);
}

__global__ void cvt_f32_bf16(const float* __restrict__ src, __bf16* __restrict__ dst, size_t n)
{
    size_t i = (size_t)blockIdx.x * blockDim.x + threadIdx.x;
    if (i < n) dst[i] = f2bf(src[i]);
}

__global__ void write_out(const float* __restrict__ ft, const float* __restrict__ fr,
                          float* __restrict__ out)
{
    int i = threadIdx.x;
    if (i < 24) {
        int b = i / 6, j = i - b * 6;
        out[i] = (j < 3) ? ft[b * 16 + j] : fr[b * 16 + (j - 3)];
    }
}

// =====================================================================
// Host-side orchestration
// =====================================================================
static inline int blk(long long threads) { return (int)((threads + 255) / 256); }

extern "C" void kernel_launch(void* const* d_in, const int* in_sizes, int n_in,
                              void* d_out, int out_size, void* d_ws, size_t ws_size,
                              hipStream_t stream)
{
    (void)in_sizes; (void)n_in; (void)out_size; (void)ws_size;
    const float* image_embs   = (const float*)d_in[0];
    const float* patch_embs   = (const float*)d_in[1];
    const float* patch_coords = (const float*)d_in[2];
    const float* fuse_w = (const float*)d_in[3];
    const float* fuse_b = (const float*)d_in[4];
    const float* ln_g   = (const float*)d_in[5];
    const float* ln_b   = (const float*)d_in[6];
    const float* d1w = (const float*)d_in[7];  const float* d1b = (const float*)d_in[8];
    const float* d2w = (const float*)d_in[9];  const float* d2b = (const float*)d_in[10];
    const float* pjw = (const float*)d_in[11]; const float* pjb = (const float*)d_in[12];
    const float* c0w = (const float*)d_in[13]; const float* c0b = (const float*)d_in[14];
    const float* c1w = (const float*)d_in[15]; const float* c1b = (const float*)d_in[16];
    const float* c2w = (const float*)d_in[17]; const float* c2b = (const float*)d_in[18];
    const float* l1w = (const float*)d_in[19]; const float* l1b = (const float*)d_in[20];
    const float* l2w = (const float*)d_in[21]; const float* l2b = (const float*)d_in[22];
    const float* l3w = (const float*)d_in[23]; const float* l3b = (const float*)d_in[24];
    const float* l4w = (const float*)d_in[25]; const float* l4b = (const float*)d_in[26];
    const float* fcw[2][3] = {{(const float*)d_in[27], (const float*)d_in[29], (const float*)d_in[31]},
                              {(const float*)d_in[33], (const float*)d_in[35], (const float*)d_in[37]}};
    const float* fcb[2][3] = {{(const float*)d_in[28], (const float*)d_in[30], (const float*)d_in[32]},
                              {(const float*)d_in[34], (const float*)d_in[36], (const float*)d_in[38]}};

    const int B = 4, D = 256, H = 192, W = 128;
    const size_t M_FUSE = 8192;                  // B*N
    const size_t PIX = (size_t)B * H * W;        // 98304

    // ---- workspace carving ----
    char* base = (char*)d_ws; size_t off = 0;
    auto carve = [&](size_t bytes) { size_t r = off; off += (bytes + 255) & ~(size_t)255; return (void*)(base + r); };
    __bf16* A_fuse   = (__bf16*)carve(M_FUSE * 512 * 2);
    __bf16* Wfuse    = (__bf16*)carve(512 * 256 * 2);
    float*  fusedPre = (float*)carve(M_FUSE * D * 4);
    float*  fused    = (float*)carve(M_FUSE * D * 4);
    float*  fm       = (float*)carve(PIX * D * 4);      // later aliased by act1/act2
    float*  wm       = (float*)carve(PIX * 4);
    __bf16* act0     = (__bf16*)carve(PIX * D * 2);
    __bf16* act1     = (__bf16*)fm;                      // alias: fm dead after splat_norm
    __bf16* act2     = (__bf16*)((char*)fm + PIX * D * 2);
    __bf16* w1t      = (__bf16*)carve(2304 * 256 * 2);
    __bf16* w2t      = (__bf16*)carve(2304 * 256 * 2);
    __bf16* wpj      = (__bf16*)carve(256 * 32 * 2);
    float*  bpj      = (float*)carve(32 * 4);
    __bf16* actP     = (__bf16*)carve(PIX * 32 * 2);
    __bf16* wc0      = (__bf16*)carve(288 * 32 * 2);
    __bf16* wc1      = (__bf16*)carve(288 * 32 * 2);
    __bf16* wc2      = (__bf16*)carve(288 * 32 * 2);
    __bf16* wl1      = (__bf16*)carve(288 * 64 * 2);
    __bf16* wl2      = (__bf16*)carve(576 * 128 * 2);
    __bf16* wl3      = (__bf16*)carve(1152 * 256 * 2);
    __bf16* wl4      = (__bf16*)carve(2304 * 256 * 2);
    __bf16* regA     = (__bf16*)carve((size_t)24576 * 32 * 2);
    __bf16* regB     = (__bf16*)carve((size_t)24576 * 32 * 2);
    __bf16* wfc1[2]  = {(__bf16*)carve(6144 * 128 * 2), (__bf16*)carve(6144 * 128 * 2)};
    __bf16* wfc2[2]  = {(__bf16*)carve(128 * 32 * 2),   (__bf16*)carve(128 * 32 * 2)};
    __bf16* wfc3[2]  = {(__bf16*)carve(32 * 16 * 2),    (__bf16*)carve(32 * 16 * 2)};
    float*  bfc3[2]  = {(float*)carve(16 * 4),          (float*)carve(16 * 4)};
    __bf16* A1       = (__bf16*)carve(16 * 6144 * 2);
    float*  o1       = (float*)carve(16 * 128 * 4);
    __bf16* A2       = (__bf16*)carve(16 * 128 * 2);
    float*  o2       = (float*)carve(16 * 32 * 4);
    __bf16* A3       = (__bf16*)carve(16 * 32 * 2);
    float*  o3[2]    = {(float*)carve(16 * 16 * 4), (float*)carve(16 * 16 * 4)};

    auto gemm1 = [&](const __bf16* A, const __bf16* Wt, const float* bias, float* out,
                     int M, int K, int N, int relu) {
        long long waves = (long long)(M / 16) * (N / 16);
        gemm_wmma_bf16<<<blk(waves * 32), 256, 0, stream>>>(A, Wt, bias, out, M, K, N, relu);
    };
    auto gemm2 = [&](const __bf16* A, const __bf16* Wt, const float* bias, float* out,
                     int M, int K, int N, int relu) {
        long long waves = (long long)(M / 32) * (N / 32);
        gemm_wmma_bf16_2x2<<<blk(waves * 32), 256, 0, stream>>>(A, Wt, bias, out, M, K, N, relu);
    };
    auto conv = [&](const __bf16* in, const __bf16* wt, const float* bias, __bf16* out,
                    int Hin, int Win, int Cin, int Ho, int Wo, int Cout,
                    int stride, int pad, int kh, int kw, int relu) {
        long long waves = (long long)((B * Ho * Wo) / 32) * (Cout / 32);
        conv_wmma<<<blk(waves * 32), 256, 0, stream>>>(in, wt, bias, out, B, Hin, Win, Cin,
                                                       Ho, Wo, Cout, stride, pad, kh, kw, relu);
    };

    // ---- 1) fuse GEMM + relu + LN ----
    build_fuseA<<<blk(M_FUSE * 512), 256, 0, stream>>>(patch_embs, image_embs, A_fuse);
    wfc_trans<<<blk(512 * 256), 256, 0, stream>>>(fuse_w, Wfuse, 512, 256, 256, 0, 0);
    gemm2(A_fuse, Wfuse, fuse_b, fusedPre, (int)M_FUSE, 512, 256, 1);
    layernorm_k<<<(int)M_FUSE, 256, 0, stream>>>(fusedPre, ln_g, ln_b, fused, D);

    // ---- 2) splat ----
    zero_f32<<<blk(PIX * D), 256, 0, stream>>>(fm, PIX * D);
    zero_f32<<<blk(PIX), 256, 0, stream>>>(wm, PIX);
    splat_scatter<<<(int)M_FUSE, 256, 0, stream>>>(fused, patch_coords, fm, wm);
    splat_norm<<<blk(PIX * D), 256, 0, stream>>>(fm, wm, act0);

    // ---- 3) dense convs + proj ----
    wconv_trans<<<blk(2304 * 256), 256, 0, stream>>>(d1w, w1t, 256, 256, 3, 3, 256, 256);
    wconv_trans<<<blk(2304 * 256), 256, 0, stream>>>(d2w, w2t, 256, 256, 3, 3, 256, 256);
    wconv_trans<<<blk(256 * 32), 256, 0, stream>>>(pjw, wpj, 4, 256, 1, 1, 256, 32);
    pad_bias<<<1, 256, 0, stream>>>(pjb, bpj, 4, 32);
    conv(act0, w1t, d1b, act1, H, W, 256, H, W, 256, 1, 1, 3, 3, 1);
    conv(act1, w2t, d2b, act2, H, W, 256, H, W, 256, 1, 1, 3, 3, 1);
    conv(act2, wpj, bpj, actP, H, W, 256, H, W, 32, 1, 0, 1, 1, 0);

    // ---- 4) regressor conv trunk ----
    wconv_trans<<<blk(288 * 32), 256, 0, stream>>>(c0w, wc0, 32, 4, 3, 3, 32, 32);
    wconv_trans<<<blk(288 * 32), 256, 0, stream>>>(c1w, wc1, 32, 32, 3, 3, 32, 32);
    wconv_trans<<<blk(288 * 32), 256, 0, stream>>>(c2w, wc2, 32, 32, 3, 3, 32, 32);
    wconv_trans<<<blk(288 * 64), 256, 0, stream>>>(l1w, wl1, 64, 32, 3, 3, 32, 64);
    wconv_trans<<<blk(576 * 128), 256, 0, stream>>>(l2w, wl2, 128, 64, 3, 3, 64, 128);
    wconv_trans<<<blk(1152 * 256), 256, 0, stream>>>(l3w, wl3, 256, 128, 3, 3, 128, 256);
    wconv_trans<<<blk(2304 * 256), 256, 0, stream>>>(l4w, wl4, 256, 256, 3, 3, 256, 256);

    conv(actP, wc0, c0b, regA, 192, 128, 32,  96, 64, 32,  2, 1, 3, 3, 1);
    conv(regA, wc1, c1b, regB,  96,  64, 32,  96, 64, 32,  1, 1, 3, 3, 1);
    conv(regB, wc2, c2b, regA,  96,  64, 32,  96, 64, 32,  1, 1, 3, 3, 1);
    conv(regA, wl1, l1b, regB,  96,  64, 32,  48, 32, 64,  2, 1, 3, 3, 1);
    conv(regB, wl2, l2b, regA,  48,  32, 64,  24, 16, 128, 2, 1, 3, 3, 1);
    conv(regA, wl3, l3b, regB,  24,  16, 128, 12,  8, 256, 2, 1, 3, 3, 1);
    conv(regB, wl4, l4b, regA,  12,   8, 256,  6,  4, 256, 2, 1, 3, 3, 1);

    // ---- 5) FC heads (M padded 4->16; fc1 K permuted NCHW->NHWC; fc3 N padded 3->16) ----
    packA_bf16<<<blk((size_t)16 * 6144), 256, 0, stream>>>(regA, A1, 4, 6144);
    for (int hsel = 0; hsel < 2; ++hsel) {
        wfc_trans<<<blk(6144 * 128), 256, 0, stream>>>(fcw[hsel][0], wfc1[hsel], 6144, 128, 128, 256, 24);
        wfc_trans<<<blk(128 * 32), 256, 0, stream>>>(fcw[hsel][1], wfc2[hsel], 128, 32, 32, 0, 0);
        wfc_trans<<<blk(32 * 16), 256, 0, stream>>>(fcw[hsel][2], wfc3[hsel], 32, 3, 16, 0, 0);
        pad_bias<<<1, 256, 0, stream>>>(fcb[hsel][2], bfc3[hsel], 3, 16);

        gemm1(A1, wfc1[hsel], fcb[hsel][0], o1, 16, 6144, 128, 1);
        cvt_f32_bf16<<<blk(16 * 128), 256, 0, stream>>>(o1, A2, (size_t)16 * 128);
        gemm1(A2, wfc2[hsel], fcb[hsel][1], o2, 16, 128, 32, 1);
        cvt_f32_bf16<<<blk(16 * 32), 256, 0, stream>>>(o2, A3, (size_t)16 * 32);
        gemm1(A3, wfc3[hsel], bfc3[hsel], o3[hsel], 16, 32, 16, 0);
    }
    write_out<<<1, 32, 0, stream>>>(o3[0], o3[1], (float*)d_out);
}
